// Vanilla_Attention_1632087573105
// MI455X (gfx1250) — matrix operand order repaired and checked by reading, not compiled, not run on hardware
//
#include <hip/hip_runtime.h>

// ============================================================================
// Vanilla attention (B=8, S=2048, E=1024) for MI455X / gfx1250.
// Staged WMMA-f16 pipeline: proj(Q,K,V) -> QK^T -> softmax -> PV.
// 160 MB of f16 intermediates in d_ws (mostly resident in the 192 MB L2).
// Round 3: fix async builtin arg types (int4 as(1)* / int4 as(3)*) per the
// compiler diagnostic; otherwise identical to round 2 (64x64 wave tiles,
// double-buffered LDS, incremental addressing, prefetch, async staging).
// ============================================================================

typedef _Float16 v8h  __attribute__((ext_vector_type(8)));
typedef _Float16 v16h __attribute__((ext_vector_type(16)));
typedef float    v8f  __attribute__((ext_vector_type(8)));

#define BM 128   // workgroup tile M
#define BN 256   // workgroup tile N
#define BK 32    // K step

#if defined(__gfx1250__) && __has_builtin(__builtin_amdgcn_global_load_async_to_lds_b128)
#define HAVE_ASYNC_LDS 1
#else
#define HAVE_ASYNC_LDS 0
#endif

__device__ __forceinline__ v8f wmma_f16(v16h a, v16h b, v8f c) {
  return __builtin_amdgcn_wmma_f32_16x16x32_f16(false, a, false, b,
                                                (short)0, c, false, false);
}

// A fragment (16x32 f16) from LDS tile sA[row][k], row stride BK.
__device__ __forceinline__ v16h frag_a(const _Float16* sA, int mOff, int lane) {
  const int r  = lane & 15;
  const int hi = lane >> 4;
  return *(const v16h*)(sA + (size_t)(mOff + r) * BK + hi * 16);
}

// B fragment (32x16 f16) from LDS tile sB[n][k] (K-major per N row).
__device__ __forceinline__ v16h frag_b(const _Float16* sB, int nOff, int lane) {
  const int n  = lane & 15;
  const int hi = lane >> 4;
  return *(const v16h*)(sB + (size_t)(nOff + n) * BK + hi * 16);
}

// 32-byte global->LDS copy of f16 data (async on gfx1250 when available).
__device__ __forceinline__ void copy32_g2l(_Float16* ldst, const _Float16* gsrc) {
#if HAVE_ASYNC_LDS
  typedef int async_v4i __attribute__((vector_size(16)));
  typedef __attribute__((address_space(1))) async_v4i ga_v4i;  // global int4
  typedef __attribute__((address_space(3))) async_v4i ls_v4i;  // LDS int4
  __builtin_amdgcn_global_load_async_to_lds_b128((ga_v4i*)gsrc,       (ls_v4i*)ldst,       0, 0);
  __builtin_amdgcn_global_load_async_to_lds_b128((ga_v4i*)(gsrc + 8), (ls_v4i*)(ldst + 8), 0, 0);
#else
  *(v16h*)ldst = *(const v16h*)gsrc;
#endif
}

__device__ __forceinline__ void async_fence() {
#if HAVE_ASYNC_LDS
#if __has_builtin(__builtin_amdgcn_s_wait_asynccnt)
  __builtin_amdgcn_s_wait_asynccnt(0);
#else
  asm volatile("s_wait_asynccnt 0" ::: "memory");
#endif
#endif
}

// Load 16 consecutive f32 and pack to v16h.
__device__ __forceinline__ v16h pack16_f32(const float* p) {
  const float4* q = (const float4*)p;
  float4 a = q[0], b = q[1], c = q[2], d = q[3];
  v16h h;
  h[0]=(_Float16)a.x;  h[1]=(_Float16)a.y;  h[2]=(_Float16)a.z;  h[3]=(_Float16)a.w;
  h[4]=(_Float16)b.x;  h[5]=(_Float16)b.y;  h[6]=(_Float16)b.z;  h[7]=(_Float16)b.w;
  h[8]=(_Float16)c.x;  h[9]=(_Float16)c.y;  h[10]=(_Float16)c.z; h[11]=(_Float16)c.w;
  h[12]=(_Float16)d.x; h[13]=(_Float16)d.y; h[14]=(_Float16)d.z; h[15]=(_Float16)d.w;
  return h;
}

// ---------------------------------------------------------------------------
// Kernel 1: Y_f16[m][n] = f16( sum_k X_f32[m][k]*W_f32[n][k] + bias[n] )
// ---------------------------------------------------------------------------
__global__ __launch_bounds__(256)
void proj_kernel(const float* __restrict__ X, const float* __restrict__ W,
                 const float* __restrict__ bias, _Float16* __restrict__ Y,
                 int N, int K) {
  __shared__ alignas(32) _Float16 sA[2][BM * BK];
  __shared__ alignas(32) _Float16 sB[2][BN * BK];

  const int tid   = threadIdx.x;
  const int lane  = tid & 31;
  const int wid   = tid >> 5;
  const int tileM = blockIdx.y * BM;
  const int tileN = blockIdx.x * BN;
  const int wm    = (wid >> 2) * 64;   // wave M: 0/64
  const int wn    = (wid & 3) * 64;    // wave N: 0/64/128/192
  const int srow  = tid >> 1;          // A staging row 0..127
  const int shalf = (tid & 1) * 16;    // A staging K half

  const float* gA = X + (size_t)(tileM + srow) * K + shalf;
  const float* gB = W + (size_t)(tileN + tid) * K;

  v8f acc[4][4];
  const v8f vzero = {0.f,0.f,0.f,0.f,0.f,0.f,0.f,0.f};
#pragma unroll
  for (int mi = 0; mi < 4; ++mi)
#pragma unroll
    for (int ni = 0; ni < 4; ++ni) acc[mi][ni] = vzero;

  // Prologue: stage K-step 0 into buffer 0.
  *(v16h*)(sA[0] + srow * BK + shalf) = pack16_f32(gA);
  *(v16h*)(sB[0] + tid * BK +  0)     = pack16_f32(gB);
  *(v16h*)(sB[0] + tid * BK + 16)     = pack16_f32(gB + 16);
  __syncthreads();

  const int nk = K / BK;
  for (int kt = 0; kt < nk; ++kt) {
    const int cur = kt & 1, nxt = cur ^ 1;
    if (kt + 1 < nk) {
      const float* a1 = gA + (size_t)(kt + 1) * BK;
      const float* b1 = gB + (size_t)(kt + 1) * BK;
      *(v16h*)(sA[nxt] + srow * BK + shalf) = pack16_f32(a1);
      *(v16h*)(sB[nxt] + tid * BK +  0)     = pack16_f32(b1);
      *(v16h*)(sB[nxt] + tid * BK + 16)     = pack16_f32(b1 + 16);
      if (kt + 2 < nk) {
        __builtin_prefetch(a1 + BK, 0, 3);
        __builtin_prefetch(b1 + BK, 0, 3);
      }
    }
    const _Float16* sAc = sA[cur];
    const _Float16* sBc = sB[cur];
    v16h af[4];
#pragma unroll
    for (int mi = 0; mi < 4; ++mi) af[mi] = frag_a(sAc, wm + mi * 16, lane);
#pragma unroll
    for (int ni = 0; ni < 4; ++ni) {
      v16h b = frag_b(sBc, wn + ni * 16, lane);
#pragma unroll
      for (int mi = 0; mi < 4; ++mi) acc[mi][ni] = wmma_f16(af[mi], b, acc[mi][ni]);
    }
    __syncthreads();
  }

  const int rowHi = 8 * (lane >> 4);
#pragma unroll
  for (int ni = 0; ni < 4; ++ni) {
    const int col = tileN + wn + ni * 16 + (lane & 15);
    const float bb = bias[col];
#pragma unroll
    for (int mi = 0; mi < 4; ++mi) {
      const int rbase = tileM + wm + mi * 16 + rowHi;
#pragma unroll
      for (int i = 0; i < 8; ++i)
        Y[(size_t)(rbase + i) * N + col] = (_Float16)(acc[mi][ni][i] + bb);
    }
  }
}

// ---------------------------------------------------------------------------
// Kernel 2: P_f16[b][q][k] = f16( scale * sum_e Qh[b][q][e]*Kh[b][k][e] )
// Pure byte-copy staging => async global->LDS path when available.
// ---------------------------------------------------------------------------
__global__ __launch_bounds__(256)
void scores_kernel(const _Float16* __restrict__ Qh, const _Float16* __restrict__ Kh,
                   _Float16* __restrict__ P, int S, int E, float scale) {
  __shared__ alignas(32) _Float16 sA[2][BM * BK];
  __shared__ alignas(32) _Float16 sB[2][BN * BK];

  const int tid   = threadIdx.x;
  const int lane  = tid & 31;
  const int wid   = tid >> 5;
  const int tileM = blockIdx.y * BM;
  const int tileN = blockIdx.x * BN;
  const int wm    = (wid >> 2) * 64;
  const int wn    = (wid & 3) * 64;
  const int srow  = tid >> 1;
  const int shalf = (tid & 1) * 16;

  const _Float16* Qb = Qh + (size_t)blockIdx.z * S * E;
  const _Float16* Kb = Kh + (size_t)blockIdx.z * S * E;
  _Float16*       Pb = P  + (size_t)blockIdx.z * S * S;

  const _Float16* gA = Qb + (size_t)(tileM + srow) * E + shalf;
  const _Float16* gB = Kb + (size_t)(tileN + tid) * E;

  v8f acc[4][4];
  const v8f vzero = {0.f,0.f,0.f,0.f,0.f,0.f,0.f,0.f};
#pragma unroll
  for (int mi = 0; mi < 4; ++mi)
#pragma unroll
    for (int ni = 0; ni < 4; ++ni) acc[mi][ni] = vzero;

  copy32_g2l(sA[0] + srow * BK + shalf, gA);
  copy32_g2l(sB[0] + tid * BK +  0, gB);
  copy32_g2l(sB[0] + tid * BK + 16, gB + 16);
  async_fence();
  __syncthreads();

  const int nk = E / BK;
  for (int kt = 0; kt < nk; ++kt) {
    const int cur = kt & 1, nxt = cur ^ 1;
    if (kt + 1 < nk) {
      const _Float16* a1 = gA + (size_t)(kt + 1) * BK;
      const _Float16* b1 = gB + (size_t)(kt + 1) * BK;
      copy32_g2l(sA[nxt] + srow * BK + shalf, a1);
      copy32_g2l(sB[nxt] + tid * BK +  0, b1);
      copy32_g2l(sB[nxt] + tid * BK + 16, b1 + 16);
      if (kt + 2 < nk) {
        __builtin_prefetch(a1 + BK, 0, 3);
        __builtin_prefetch(b1 + BK, 0, 3);
      }
    }
    const _Float16* sAc = sA[cur];
    const _Float16* sBc = sB[cur];
    v16h af[4];
#pragma unroll
    for (int mi = 0; mi < 4; ++mi) af[mi] = frag_a(sAc, wm + mi * 16, lane);
#pragma unroll
    for (int ni = 0; ni < 4; ++ni) {
      v16h b = frag_b(sBc, wn + ni * 16, lane);
#pragma unroll
      for (int mi = 0; mi < 4; ++mi) acc[mi][ni] = wmma_f16(af[mi], b, acc[mi][ni]);
    }
    async_fence();
    __syncthreads();
  }

  const int rowHi = 8 * (lane >> 4);
#pragma unroll
  for (int ni = 0; ni < 4; ++ni) {
    const int col = tileN + wn + ni * 16 + (lane & 15);
#pragma unroll
    for (int mi = 0; mi < 4; ++mi) {
      const int rbase = tileM + wm + mi * 16 + rowHi;
#pragma unroll
      for (int i = 0; i < 8; ++i)
        Pb[(size_t)(rbase + i) * S + col] = (_Float16)(acc[mi][ni][i] * scale);
    }
  }
}

// ---------------------------------------------------------------------------
// Kernel 3: in-place row softmax over f16 rows of length S (one block / row).
// ---------------------------------------------------------------------------
__global__ __launch_bounds__(256)
void softmax_kernel(_Float16* __restrict__ P, int S) {
  __shared__ float red[8];
  const int tid  = threadIdx.x;
  const int lane = tid & 31;
  const int wid  = tid >> 5;
  _Float16* row = P + (size_t)blockIdx.x * S;

  v8h xh = *(const v8h*)(row + tid * 8);
  float x[8];
  float m = -1e30f;
#pragma unroll
  for (int i = 0; i < 8; ++i) { x[i] = (float)xh[i]; m = fmaxf(m, x[i]); }
#pragma unroll
  for (int off = 16; off; off >>= 1) m = fmaxf(m, __shfl_xor(m, off, 32));
  if (lane == 0) red[wid] = m;
  __syncthreads();
  float gm = red[0];
#pragma unroll
  for (int i = 1; i < 8; ++i) gm = fmaxf(gm, red[i]);
  __syncthreads();

  float s = 0.f;
#pragma unroll
  for (int i = 0; i < 8; ++i) { x[i] = __expf(x[i] - gm); s += x[i]; }
#pragma unroll
  for (int off = 16; off; off >>= 1) s += __shfl_xor(s, off, 32);
  if (lane == 0) red[wid] = s;
  __syncthreads();
  float gs = 0.f;
#pragma unroll
  for (int i = 0; i < 8; ++i) gs += red[i];
  const float inv = 1.0f / gs;

  v8h o;
#pragma unroll
  for (int i = 0; i < 8; ++i) o[i] = (_Float16)(x[i] * inv);
  *(v8h*)(row + tid * 8) = o;
}

// ---------------------------------------------------------------------------
// Kernel 4: O_f32[b][q][e] = sum_k P[b][q][k] * Vh[b][k][e]
// A tile: direct copy (async-capable). B tile: transpose via DS scatter.
// ---------------------------------------------------------------------------
__global__ __launch_bounds__(256)
void pv_kernel(const _Float16* __restrict__ P, const _Float16* __restrict__ Vh,
               float* __restrict__ O, int S, int E) {
  __shared__ alignas(32) _Float16 sA[2][BM * BK];
  __shared__ alignas(32) _Float16 sB[2][BN * BK];

  const int tid   = threadIdx.x;
  const int lane  = tid & 31;
  const int wid   = tid >> 5;
  const int tileM = blockIdx.y * BM;   // q rows
  const int tileN = blockIdx.x * BN;   // e cols
  const int wm    = (wid >> 2) * 64;
  const int wn    = (wid & 3) * 64;
  const int srow  = tid >> 1;
  const int shalf = (tid & 1) * 16;
  const int vk    = tid >> 3;          // 0..31  key within tile
  const int ve    = (tid & 7) * 32;    // 0..224 e chunk within tile

  const _Float16* Pb = P  + (size_t)blockIdx.z * S * S;
  const _Float16* Vb = Vh + (size_t)blockIdx.z * S * E;
  float*          Ob = O  + (size_t)blockIdx.z * S * E;

  const _Float16* gA = Pb + (size_t)(tileM + srow) * S + shalf;
  const _Float16* gV = Vb + (size_t)vk * E + tileN + ve;

  v8f acc[4][4];
  const v8f vzero = {0.f,0.f,0.f,0.f,0.f,0.f,0.f,0.f};
#pragma unroll
  for (int mi = 0; mi < 4; ++mi)
#pragma unroll
    for (int ni = 0; ni < 4; ++ni) acc[mi][ni] = vzero;

  // Prologue staging (buffer 0).
  copy32_g2l(sA[0] + srow * BK + shalf, gA);
  {
    v16h v0 = *(const v16h*)(gV);
    v16h v1 = *(const v16h*)(gV + 16);
#pragma unroll
    for (int i = 0; i < 16; ++i) sB[0][(size_t)(ve + i) * BK + vk]      = v0[i];
#pragma unroll
    for (int i = 0; i < 16; ++i) sB[0][(size_t)(ve + 16 + i) * BK + vk] = v1[i];
  }
  async_fence();
  __syncthreads();

  const int nk = S / BK;   // K dim = keys
  for (int kt = 0; kt < nk; ++kt) {
    const int cur = kt & 1, nxt = cur ^ 1;
    if (kt + 1 < nk) {
      const _Float16* a1  = gA + (size_t)(kt + 1) * BK;
      const _Float16* v1p = gV + (size_t)(kt + 1) * BK * E;
      copy32_g2l(sA[nxt] + srow * BK + shalf, a1);
      v16h v0 = *(const v16h*)(v1p);
      v16h v1 = *(const v16h*)(v1p + 16);
#pragma unroll
      for (int i = 0; i < 16; ++i) sB[nxt][(size_t)(ve + i) * BK + vk]      = v0[i];
#pragma unroll
      for (int i = 0; i < 16; ++i) sB[nxt][(size_t)(ve + 16 + i) * BK + vk] = v1[i];
      if (kt + 2 < nk) {
        __builtin_prefetch(a1 + BK, 0, 3);
        __builtin_prefetch(v1p + (size_t)BK * E, 0, 3);
      }
    }
    const _Float16* sAc = sA[cur];
    const _Float16* sBc = sB[cur];
    v16h af[4];
#pragma unroll
    for (int mi = 0; mi < 4; ++mi) af[mi] = frag_a(sAc, wm + mi * 16, lane);
#pragma unroll
    for (int ni = 0; ni < 4; ++ni) {
      v16h b = frag_b(sBc, wn + ni * 16, lane);
#pragma unroll
      for (int mi = 0; mi < 4; ++mi) acc[mi][ni] = wmma_f16(af[mi], b, acc[mi][ni]);
    }
    async_fence();
    __syncthreads();
  }

  const int rowHi = 8 * (lane >> 4);
#pragma unroll
  for (int ni = 0; ni < 4; ++ni) {
    const int col = tileN + wn + ni * 16 + (lane & 15);
#pragma unroll
    for (int mi = 0; mi < 4; ++mi) {
      const int rbase = tileM + wm + mi * 16 + rowHi;
#pragma unroll
      for (int i = 0; i < 8; ++i)
        Ob[(size_t)(rbase + i) * E + col] = acc[mi][ni][i];
    }
  }
}

// ---------------------------------------------------------------------------
extern "C" void kernel_launch(void* const* d_in, const int* in_sizes, int n_in,
                              void* d_out, int out_size, void* d_ws, size_t ws_size,
                              hipStream_t stream) {
  (void)in_sizes; (void)n_in; (void)out_size; (void)ws_size;

  constexpr int B = 8, S = 2048, E = 1024;
  const size_t BSE = (size_t)B * S * E;

  const float* q  = (const float*)d_in[0];
  const float* k  = (const float*)d_in[1];
  const float* v  = (const float*)d_in[2];
  const float* Wq = (const float*)d_in[3];
  const float* bq = (const float*)d_in[4];
  const float* Wk = (const float*)d_in[5];
  const float* bk = (const float*)d_in[6];
  const float* Wv = (const float*)d_in[7];
  const float* bv = (const float*)d_in[8];

  // Workspace layout (f16): Qh | Kh | Vh | P   (32+32+32+64 = 160 MB)
  _Float16* Qh = (_Float16*)d_ws;
  _Float16* Kh = Qh + BSE;
  _Float16* Vh = Kh + BSE;
  _Float16* P  = Vh + BSE;

  const dim3 blk(256);
  const dim3 gProj(E / BN, (B * S) / BM);                 // (4, 128)
  proj_kernel<<<gProj, blk, 0, stream>>>(q, Wq, bq, Qh, E, E);
  proj_kernel<<<gProj, blk, 0, stream>>>(k, Wk, bk, Kh, E, E);
  proj_kernel<<<gProj, blk, 0, stream>>>(v, Wv, bv, Vh, E, E);

  const float scale = 0.03125f;                           // 1/sqrt(1024)
  scores_kernel<<<dim3(S / BN, S / BM, B), blk, 0, stream>>>(Qh, Kh, P, S, E, scale);

  softmax_kernel<<<dim3(B * S), blk, 0, stream>>>(P, S);

  pv_kernel<<<dim3(E / BN, S / BM, B), blk, 0, stream>>>(P, Vh, (float*)d_out, S, E);
}
